// CustomLSTM_32040456028222
// MI455X (gfx1250) — compile-verified, matrix operand order checked by reading, not run on hardware
//
#include <hip/hip_runtime.h>

// ---------------------------------------------------------------------------
// CDNA5 (gfx1250) LSTM:
//   bf16 WMMA GEMMs with TDM (tensor_load_to_lds) double-buffered LDS staging
//   + fused per-timestep recurrence.
// ---------------------------------------------------------------------------

typedef __attribute__((ext_vector_type(16))) __bf16        v16bf;
typedef __attribute__((ext_vector_type(8)))  float         v8f;
typedef __attribute__((ext_vector_type(4)))  unsigned int  v4u;
typedef __attribute__((ext_vector_type(8)))  int           v8i;
typedef __attribute__((ext_vector_type(4)))  int           v4i;

static constexpr int B_ = 64, T_ = 512, R_ = 256, E_ = 512, H_ = 1024;
static constexpr int BT_ = B_ * T_;
static constexpr int G_  = 4 * H_;

// TDM pad feature: after every 16 dwords (32 bf16) insert 4 dwords (8 bf16)
// -> LDS row stride 40 bf16 (80B), staggering rows across the 64 banks.
static constexpr int LDS_ROW = 40;

union Frag16 {
    v16bf v;
    uint4 q[2];
};

__device__ __forceinline__ unsigned short f2bf(float f) {
    union { float f; unsigned u; } a; a.f = f;
    unsigned u = a.u + 0x7FFFu + ((a.u >> 16) & 1u);   // round-to-nearest-even
    return (unsigned short)(u >> 16);
}

// 16x32 bf16 fragment slice for this lane (ISA 7.12.2): two 16B chunks,
// K {off..off+7} and {off+16..off+23}; p = row base + 8*laneHi elems.
__device__ __forceinline__ v16bf load_frag_lds(const unsigned short* __restrict__ p) {
    Frag16 f;
    const uint4* q = reinterpret_cast<const uint4*>(p);   // ds_load_b128 x2
    f.q[0] = q[0];
    f.q[1] = q[2];
    return f.v;
}

__device__ __forceinline__ float sigmoidf_(float x) {
    return 1.0f / (1.0f + __expf(-x));
}

// ---------------------------------------------------------------------------
// TDM: issue one tensor_load_to_lds with a hand-built D# (ISA ch.8).
//   Tile: tile0 (contiguous, elems) x tile1 (rows) x tile2 (planes).
//   s0 = row stride (elems), s1 = plane stride (elems). data_size = 2 bytes.
//   Pad: 4 dwords after every 16 dwords -> LDS row stride = LDS_ROW bf16.
// ---------------------------------------------------------------------------
__device__ __forceinline__ void tdm_load(
    unsigned lds_addr, const void* gptr,
    unsigned td0, unsigned td1, unsigned td2,
    unsigned tile0, unsigned tile1, unsigned tile2,
    unsigned long long s0, unsigned long long s1) {

    unsigned long long ga = (unsigned long long)(size_t)gptr;

    v4u g0;
    g0[0] = 1u;                                            // count=1, user desc
    g0[1] = lds_addr;                                      // LDS byte address
    g0[2] = (unsigned)(ga & 0xFFFFFFFFull);                // global_addr[31:0]
    g0[3] = (unsigned)((ga >> 32) & 0x01FFFFFFull)         // global_addr[56:32]
          | (2u << 30);                                    // type=2 (image)

    v8i g1;
    g1[0] = (int)((1u << 16)        // data_size = 1 -> 2-byte elements
                | (1u << 20)        // pad_enable
                | (3u << 22)        // pad_interval: 16 dwords
                | (3u << 25));      // pad_amount:   4 dwords
    g1[1] = (int)((td0 & 0xFFFFu) << 16);                            // dim0[15:0]
    g1[2] = (int)((td0 >> 16) | ((td1 & 0xFFFFu) << 16));            // dim0[31:16], dim1[15:0]
    g1[3] = (int)((td1 >> 16) | (tile0 << 16));                      // dim1[31:16], tile0
    g1[4] = (int)(tile1 | (tile2 << 16));                            // tile1, tile2
    g1[5] = (int)(s0 & 0xFFFFFFFFull);                               // stride0[31:0]
    g1[6] = (int)((unsigned)((s0 >> 32) & 0xFFFFull)
                | ((unsigned)(s1 & 0xFFFFull) << 16));               // stride0[47:32], stride1[15:0]
    g1[7] = (int)((unsigned)((s1 >> 16) & 0xFFFFFFFFull));           // stride1[47:16]

    v4i g2 = {(int)td2, 0, 0, 0};   // tensor_dim2 (OOB for plane index)
    v4i g3 = {0, 0, 0, 0};

#if defined(__clang_major__) && (__clang_major__ >= 23)
    v8i g4 = {0, 0, 0, 0, 0, 0, 0, 0};
    __builtin_amdgcn_tensor_load_to_lds(g0, g1, g2, g3, g4, 0);
#else
    __builtin_amdgcn_tensor_load_to_lds(g0, g1, g2, g3, 0);
#endif
}

// ---------------------------------------------------------------------------
// float -> bf16 bulk conversion
// ---------------------------------------------------------------------------
__global__ __launch_bounds__(256) void f32_to_bf16_kernel(
    const float* __restrict__ in, unsigned short* __restrict__ out, size_t n) {
    size_t i = (size_t)blockIdx.x * blockDim.x + threadIdx.x;
    if (i < n) out[i] = f2bf(in[i]);
}

// ---------------------------------------------------------------------------
// zero-init h0 (bf16) and c0 (f32)
// ---------------------------------------------------------------------------
__global__ __launch_bounds__(256) void init_state_kernel(
    unsigned short* __restrict__ h0, float* __restrict__ c0, int n) {
    int i = blockIdx.x * blockDim.x + threadIdx.x;
    if (i < n) { h0[i] = 0; c0[i] = 0.0f; }
}

// ---------------------------------------------------------------------------
// C[M,N] = A[M,K](bf16) @ B[N,K](bf16)^T + bias0[N] (+ bias1[N])
// Block 256 thr = 8 waves (2M x 4N); block tile 32M x 256N; wave tile 16M x 64N.
// TDM double-buffered LDS staging: wave 0 DMAs K-tile kt+1 while all waves
// run WMMA on K-tile kt from LDS.
// ---------------------------------------------------------------------------
template <bool BF16OUT>
__global__ __launch_bounds__(256) void wmma_gemm_tdm_kernel(
    const unsigned short* __restrict__ A,
    const unsigned short* __restrict__ Bm,
    const float* __restrict__ bias0,
    const float* __restrict__ bias1,
    void* __restrict__ Cout,
    int M, int N, int K) {

    __shared__ unsigned short As[2][32  * LDS_ROW];
    __shared__ unsigned short Bs[2][256 * LDS_ROW];

    const int lane   = threadIdx.x & 31;
    const int wave   = threadIdx.x >> 5;
    const int waveM  = wave >> 2;      // 0..1
    const int waveN  = wave & 3;       // 0..3
    const int laneHi = lane >> 4;
    const int laneLo = lane & 15;

    const int mBase = blockIdx.x * 32;
    const int nBase = blockIdx.y * 256;
    const int KT    = K >> 5;

    v8f acc[4] = {};

    if (wave == 0) {   // prologue: DMA K-tile 0 into buffer 0
        tdm_load((unsigned)(size_t)&As[0][0], A  + (size_t)mBase * K,
                 (unsigned)K, 32u,  0u, 32u, 32u,  0u, (unsigned long long)K, 0ull);
        tdm_load((unsigned)(size_t)&Bs[0][0], Bm + (size_t)nBase * K,
                 (unsigned)K, 256u, 0u, 32u, 256u, 0u, (unsigned long long)K, 0ull);
    }

    for (int kt = 0; kt < KT; ++kt) {
        const int buf = kt & 1;
        if (wave == 0) {
            if (kt + 1 < KT) {
                const int nb = buf ^ 1;
                const int ko = (kt + 1) * 32;
                tdm_load((unsigned)(size_t)&As[nb][0], A  + (size_t)mBase * K + ko,
                         (unsigned)K, 32u,  0u, 32u, 32u,  0u, (unsigned long long)K, 0ull);
                tdm_load((unsigned)(size_t)&Bs[nb][0], Bm + (size_t)nBase * K + ko,
                         (unsigned)K, 256u, 0u, 32u, 256u, 0u, (unsigned long long)K, 0ull);
                __builtin_amdgcn_s_wait_tensorcnt(2);   // tile kt landed
            } else {
                __builtin_amdgcn_s_wait_tensorcnt(0);
            }
        }
        __syncthreads();   // LDS tile kt visible to all waves

        v16bf a = load_frag_lds(&As[buf][(waveM * 16 + laneLo) * LDS_ROW + 8 * laneHi]);
#pragma unroll
        for (int j = 0; j < 4; ++j) {
            v16bf b = load_frag_lds(&Bs[buf][(waveN * 64 + 16 * j + laneLo) * LDS_ROW + 8 * laneHi]);
            acc[j] = __builtin_amdgcn_wmma_f32_16x16x32_bf16(
                false, a, false, b, (short)0, acc[j], false, false);
        }
        __syncthreads();   // done reading before buf is re-filled at kt+2
    }

    // Epilogue: elem i -> row mBase+waveM*16+8*laneHi+i, col nBase+waveN*64+16j+laneLo
#pragma unroll
    for (int j = 0; j < 4; ++j) {
        const int n = nBase + waveN * 64 + 16 * j + laneLo;
        float bs = bias0[n] + (bias1 ? bias1[n] : 0.0f);
#pragma unroll
        for (int i = 0; i < 8; ++i) {
            const int m = mBase + waveM * 16 + 8 * laneHi + i;
            float v = acc[j][i] + bs;
            if (BF16OUT)
                ((unsigned short*)Cout)[(size_t)m * N + n] = f2bf(v);
            else
                ((float*)Cout)[(size_t)m * N + n] = v;
        }
    }
}

// ---------------------------------------------------------------------------
// One LSTM timestep, fully fused, TDM-staged:
//   gates[b, g*H+n] = gx[b*T+t, g*H+n] + (h_prev @ W_hh^T)[b, g*H+n]
//   c = sig(f)*c + sig(i)*tanh(g);  h = sig(o)*tanh(c)
// Wave computes one 16(batch) x 16(H) tile for ALL 4 gates (shared A frag).
// W_hh rows for all 4 gates fetched by ONE 3D TDM descriptor:
//   K(32) x rows(64) x gates(4), plane stride = H*H elems.
// Grid (B/32, H/64); h ping-pongs between two bf16 buffers across steps.
// ---------------------------------------------------------------------------
__global__ __launch_bounds__(256) void lstm_step_kernel(
    const unsigned short* __restrict__ hPrev,   // [B,H] bf16
    const unsigned short* __restrict__ Whh,     // [4H,H] bf16
    const float* __restrict__ gx,               // [B*T, 4H] f32 (biases folded in)
    float* __restrict__ c,                      // [B,H] f32, in-place
    unsigned short* __restrict__ hNext,         // [B,H] bf16
    float* __restrict__ outputs,                // [B,T,H] f32
    float* __restrict__ outH,                   // [B,H] final h
    float* __restrict__ outC,                   // [B,H] final c
    int t, int isLast) {

    __shared__ unsigned short Hs[2][32  * LDS_ROW];
    __shared__ unsigned short Ws[2][256 * LDS_ROW];   // 4 gates x 64 rows

    const int lane   = threadIdx.x & 31;
    const int wave   = threadIdx.x >> 5;
    const int waveM  = wave >> 2;
    const int waveN  = wave & 3;
    const int laneHi = lane >> 4;
    const int laneLo = lane & 15;

    const int mBase = blockIdx.x * 32;    // batch rows
    const int nBase = blockIdx.y * 64;    // H columns
    const int KT    = H_ >> 5;            // 32

    // Prefetch epilogue gx operands early (global_prefetch_b8) so the fp32
    // adds after the WMMA loop hit L2/L0.
    {
        const int n0 = nBase + waveN * 16 + laneLo;
        const int b0 = mBase + waveM * 16 + 8 * laneHi;
        const float* gp = gx + ((size_t)b0 * T_ + t) * G_ + n0;
        __builtin_prefetch(gp,          0, 1);
        __builtin_prefetch(gp + 1 * H_, 0, 1);
        __builtin_prefetch(gp + 2 * H_, 0, 1);
        __builtin_prefetch(gp + 3 * H_, 0, 1);
    }

    v8f acc[4] = {};   // gates i,f,g,o

    if (wave == 0) {
        tdm_load((unsigned)(size_t)&Hs[0][0], hPrev + (size_t)mBase * H_,
                 (unsigned)H_, 32u, 0u, 32u, 32u, 0u, (unsigned long long)H_, 0ull);
        tdm_load((unsigned)(size_t)&Ws[0][0], Whh + (size_t)nBase * H_,
                 (unsigned)H_, 64u, 4u, 32u, 64u, 4u,
                 (unsigned long long)H_, (unsigned long long)H_ * (unsigned long long)H_);
    }

    for (int kt = 0; kt < KT; ++kt) {
        const int buf = kt & 1;
        if (wave == 0) {
            if (kt + 1 < KT) {
                const int nb = buf ^ 1;
                const int ko = (kt + 1) * 32;
                tdm_load((unsigned)(size_t)&Hs[nb][0], hPrev + (size_t)mBase * H_ + ko,
                         (unsigned)H_, 32u, 0u, 32u, 32u, 0u, (unsigned long long)H_, 0ull);
                tdm_load((unsigned)(size_t)&Ws[nb][0], Whh + (size_t)nBase * H_ + ko,
                         (unsigned)H_, 64u, 4u, 32u, 64u, 4u,
                         (unsigned long long)H_, (unsigned long long)H_ * (unsigned long long)H_);
                __builtin_amdgcn_s_wait_tensorcnt(2);
            } else {
                __builtin_amdgcn_s_wait_tensorcnt(0);
            }
        }
        __syncthreads();

        v16bf a = load_frag_lds(&Hs[buf][(waveM * 16 + laneLo) * LDS_ROW + 8 * laneHi]);
#pragma unroll
        for (int g = 0; g < 4; ++g) {
            v16bf b = load_frag_lds(&Ws[buf][(g * 64 + waveN * 16 + laneLo) * LDS_ROW + 8 * laneHi]);
            acc[g] = __builtin_amdgcn_wmma_f32_16x16x32_bf16(
                false, a, false, b, (short)0, acc[g], false, false);
        }
        __syncthreads();
    }

    const int n = nBase + waveN * 16 + laneLo;
#pragma unroll
    for (int i = 0; i < 8; ++i) {
        const int b = mBase + waveM * 16 + 8 * laneHi + i;
        const size_t gRow = ((size_t)b * T_ + t) * G_;

        float gi = acc[0][i] + gx[gRow + 0 * H_ + n];
        float gf = acc[1][i] + gx[gRow + 1 * H_ + n];
        float gg = acc[2][i] + gx[gRow + 2 * H_ + n];
        float go = acc[3][i] + gx[gRow + 3 * H_ + n];

        float si = sigmoidf_(gi);
        float sf = sigmoidf_(gf);
        float tg = tanhf(gg);
        float so = sigmoidf_(go);

        const size_t cn = (size_t)b * H_ + n;
        float cNew = sf * c[cn] + si * tg;
        float hNew = so * tanhf(cNew);

        c[cn]     = cNew;
        hNext[cn] = f2bf(hNew);
        outputs[(size_t)b * (size_t)T_ * H_ + (size_t)t * H_ + n] = hNew;
        if (isLast) { outH[cn] = hNew; outC[cn] = cNew; }
    }
}

// ---------------------------------------------------------------------------
// Host launcher
// ---------------------------------------------------------------------------
extern "C" void kernel_launch(void* const* d_in, const int* in_sizes, int n_in,
                              void* d_out, int out_size, void* d_ws, size_t ws_size,
                              hipStream_t stream) {
    (void)in_sizes; (void)n_in; (void)out_size; (void)ws_size;

    const float* x     = (const float*)d_in[0];   // [B,T,R]
    const float* W_emb = (const float*)d_in[1];   // [E,R]
    const float* b_emb = (const float*)d_in[2];   // [E]
    const float* W_ih  = (const float*)d_in[3];   // [4H,E]
    const float* W_hh  = (const float*)d_in[4];   // [4H,H]
    const float* b_ih  = (const float*)d_in[5];   // [4H]
    const float* b_hh  = (const float*)d_in[6];   // [4H]

    float* out     = (float*)d_out;
    float* outputs = out;                               // [B,T,H]
    float* outH    = out + (size_t)B_ * T_ * H_;        // [B,H]
    float* outC    = outH + (size_t)B_ * H_;            // [B,H]

    // ---- workspace carve-up (256B aligned) ----
    char* ws = (char*)d_ws;
    size_t off = 0;
    auto carve = [&](size_t bytes) -> char* {
        char* p = ws + off;
        off = (off + bytes + 255) & ~(size_t)255;
        return p;
    };
    unsigned short* xbf    = (unsigned short*)carve((size_t)BT_ * R_ * 2);
    unsigned short* wembbf = (unsigned short*)carve((size_t)E_ * R_ * 2);
    unsigned short* wihbf  = (unsigned short*)carve((size_t)G_ * E_ * 2);
    unsigned short* whhbf  = (unsigned short*)carve((size_t)G_ * H_ * 2);
    unsigned short* xebf   = (unsigned short*)carve((size_t)BT_ * E_ * 2);
    float*          gxf    = (float*)carve((size_t)BT_ * G_ * 4);
    unsigned short* hbuf   = (unsigned short*)carve((size_t)2 * B_ * H_ * 2);
    float*          cbuf   = (float*)carve((size_t)B_ * H_ * 4);

    const int BH = B_ * H_;

    // ---- 1) f32 -> bf16 conversions ----
    auto cvt = [&](const float* in, unsigned short* o, size_t n) {
        int blocks = (int)((n + 255) / 256);
        f32_to_bf16_kernel<<<blocks, 256, 0, stream>>>(in, o, n);
    };
    cvt(x,     xbf,    (size_t)BT_ * R_);
    cvt(W_emb, wembbf, (size_t)E_ * R_);
    cvt(W_ih,  wihbf,  (size_t)G_ * E_);
    cvt(W_hh,  whhbf,  (size_t)G_ * H_);

    // ---- 2) xe = x @ W_emb^T + b_emb   (bf16 out) ----
    {
        dim3 grid(BT_ / 32, E_ / 256);
        wmma_gemm_tdm_kernel<true><<<grid, 256, 0, stream>>>(
            xbf, wembbf, b_emb, nullptr, (void*)xebf, BT_, E_, R_);
    }

    // ---- 3) gx = xe @ W_ih^T + (b_ih + b_hh)   (f32 out) ----
    {
        dim3 grid(BT_ / 32, G_ / 256);
        wmma_gemm_tdm_kernel<false><<<grid, 256, 0, stream>>>(
            xebf, wihbf, b_ih, b_hh, (void*)gxf, BT_, G_, E_);
    }

    // ---- 4) init h0/c0 ----
    init_state_kernel<<<(BH + 255) / 256, 256, 0, stream>>>(hbuf, cbuf, BH);

    // ---- 5) recurrence: one fused kernel per timestep ----
    for (int t = 0; t < T_; ++t) {
        const unsigned short* hP = hbuf + (size_t)(t & 1) * BH;
        unsigned short*       hN = hbuf + (size_t)((t + 1) & 1) * BH;
        dim3 grid(B_ / 32, H_ / 64);
        lstm_step_kernel<<<grid, 256, 0, stream>>>(
            hP, whhbf, gxf, cbuf, hN, outputs, outH, outC, t, t == T_ - 1);
    }
}
